// LongformerSelfAttentionMine_18107582120750
// MI455X (gfx1250) — matrix operand order repaired and checked by reading, not compiled, not run on hardware
//
#include <hip/hip_runtime.h>
#include <math.h>

#define Bsz   2
#define Ssz   2048
#define DIMsz 768
#define Hsz   12
#define DHsz  64
#define Wsz   256
#define Gsz   16

typedef __attribute__((ext_vector_type(16))) _Float16 v16h;
typedef __attribute__((ext_vector_type(8)))  float    v8f;

union HV  { v16h h; float4 f[2]; };
union HI4 { int4 i4; _Float16 h[8]; };

// ---- CDNA5 async global->LDS copy (16B per active lane), tracked by ASYNCcnt ----
__device__ __forceinline__ void async_cp16(void* lds_dst, const void* gsrc) {
  unsigned laddr = (unsigned)(unsigned long long)lds_dst;   // low 32b of generic LDS ptr
  asm volatile("global_load_async_to_lds_b128 %0, %1, off"
               :: "v"(laddr), "v"(gsrc) : "memory");
}
__device__ __forceinline__ void async_wait0() {
  asm volatile("s_wait_asynccnt 0x0" ::: "memory");
}

// A-fragment (16x32, MxK), per CDNA5 ISA layout:
// lane l (0-15) / l+16: row M=l, halves 0..7 = K hi*8..+7, halves 8..15 = K 16+hi*8..
__device__ __forceinline__ v16h load_a_frag(const _Float16* base, int stride_h, int l, int hi) {
  HV u;
  const char* p = (const char*)(base + (size_t)l * stride_h);
  u.f[0] = *(const float4*)(p + hi * 16);
  u.f[1] = *(const float4*)(p + 32 + hi * 16);
  return u.h;
}

// B-fragment (32x16, KxN) gathered from an N-major (transposed) buffer:
// lane l holds column n=l, halves 0..7 = K hi*8..+7, halves 8..15 = K 16+hi*8.. (same as A).
__device__ __forceinline__ v16h load_bt_frag(const _Float16* baseT, int stride_h, int l, int hi) {
  HV u;
  const char* p = (const char*)(baseT + (size_t)l * stride_h);
  u.f[0] = *(const float4*)(p + hi * 16);
  u.f[1] = *(const float4*)(p + 32 + hi * 16);
  return u.h;
}

__device__ __forceinline__ v8f wmma32(v16h a, v16h b, v8f c) {
  return __builtin_amdgcn_wmma_f32_16x16x32_f16(false, a, false, b, (short)0, c, false, false);
}

__device__ __forceinline__ v8f zero8() {
  v8f z = {0.f, 0.f, 0.f, 0.f, 0.f, 0.f, 0.f, 0.f};
  return z;
}

__device__ __forceinline__ HI4 cvt8(float4 f0, float4 f1) {
  HI4 o;
  o.h[0] = (_Float16)f0.x; o.h[1] = (_Float16)f0.y;
  o.h[2] = (_Float16)f0.z; o.h[3] = (_Float16)f0.w;
  o.h[4] = (_Float16)f1.x; o.h[5] = (_Float16)f1.y;
  o.h[6] = (_Float16)f1.z; o.h[7] = (_Float16)f1.w;
  return o;
}

// ---------------------------------------------------------------------------
// GEMM: Y[M,768] = ((X[M,768] @ W[768,768]) + bias) * scale, f32 in -> f16 out
// block = 256 threads (8 waves), tile 128(M) x 64(N), K step 32. K=N=768 fixed.
// ---------------------------------------------------------------------------
__global__ __launch_bounds__(256) void gemm_bias_f16(
    const float* __restrict__ X, const float* __restrict__ Wm,
    const float* __restrict__ bias, _Float16* __restrict__ Y,
    int M, float scale)
{
  constexpr int K = 768, N = 768;
  __shared__ _Float16 As[128 * 32];
  __shared__ _Float16 Bt[64 * 32];

  const int tid  = threadIdx.x;
  const int lane = tid & 31, wave = tid >> 5;
  const int l = lane & 15, hi = lane >> 4;
  const int wm = wave & 3, wn = wave >> 2;
  const int m0 = blockIdx.y * 128;
  const int n0 = blockIdx.x * 64;

  v8f acc[2][2];
  for (int i = 0; i < 2; ++i)
    for (int j = 0; j < 2; ++j) acc[i][j] = zero8();

  // A: thread -> (row, 16-col half); clamp row (out-of-range rows never stored)
  const int ar = tid >> 1, ah = (tid & 1) * 16;
  int xr = m0 + ar; if (xr >= M) xr = M - 1;
  const float* xrow = X + (size_t)xr * K;
  // B: thread -> (k-row, 8 n-cols), stored transposed
  const int bk = tid >> 3, bc = (tid & 7) * 8;

  for (int k0 = 0; k0 < K; k0 += 32) {
    {
      const float4* src = (const float4*)(xrow + k0 + ah);
      float4 f0 = src[0], f1 = src[1], f2 = src[2], f3 = src[3];
      int4* dst = (int4*)(As + ar * 32 + ah);
      dst[0] = cvt8(f0, f1).i4;
      dst[1] = cvt8(f2, f3).i4;
    }
    {
      const float* src = Wm + (size_t)(k0 + bk) * N + n0 + bc;
      float4 f0 = *(const float4*)src;
      float4 f1 = *(const float4*)(src + 4);
      HI4 u = cvt8(f0, f1);
      for (int e = 0; e < 8; ++e) Bt[(bc + e) * 32 + bk] = u.h[e];
    }
    __syncthreads();

    v16h a0 = load_a_frag(As + (wm * 32 + 0)  * 32, 32, l, hi);
    v16h a1 = load_a_frag(As + (wm * 32 + 16) * 32, 32, l, hi);
    v16h b0 = load_bt_frag(Bt + (wn * 32 + 0)  * 32, 32, l, hi);
    v16h b1 = load_bt_frag(Bt + (wn * 32 + 16) * 32, 32, l, hi);
    acc[0][0] = wmma32(a0, b0, acc[0][0]);
    acc[0][1] = wmma32(a0, b1, acc[0][1]);
    acc[1][0] = wmma32(a1, b0, acc[1][0]);
    acc[1][1] = wmma32(a1, b1, acc[1][1]);
    __syncthreads();
  }

  // Epilogue: M is a multiple of 16, so each 16-row D tile is fully in or out.
  for (int mi = 0; mi < 2; ++mi) {
    int rowbase = m0 + wm * 32 + mi * 16;
    if (rowbase >= M) continue;                       // wave-uniform
    for (int nj = 0; nj < 2; ++nj) {
      int col = n0 + wn * 32 + nj * 16 + l;
      float bv = bias[col];
      _Float16* yp = Y + (size_t)(rowbase + hi * 8) * N + col;
      for (int r = 0; r < 8; ++r)                     // C/D: VGPR r -> M = r + hi*8
        yp[(size_t)r * N] = (_Float16)((acc[mi][nj][r] + bv) * scale);
    }
  }
}

// ---------------------------------------------------------------------------
// Band (sliding window + global-key columns) attention.
// One block per (b, h, 64-query tile). Key panel: 16 global + 576 window + 16 pad.
// Out-of-range slots are CLAMPED (not zeroed): their scores are masked to -inf,
// so their probs are exactly 0 and the clamped V rows contribute nothing.
// ---------------------------------------------------------------------------
__global__ __launch_bounds__(256) void band_attn(
    const _Float16* __restrict__ q, const _Float16* __restrict__ k,
    const _Float16* __restrict__ v, const int* __restrict__ am,
    float* __restrict__ out)
{
  extern __shared__ char smem[];
  _Float16* Qs = (_Float16*)smem;          // 64 x 64
  _Float16* KV = Qs + 64 * 64;             // Kext: 608 x 64 (later VT: 64 x 608)
  _Float16* Sc = KV + 608 * 64;            // 64 x 608 scores -> probs (f16, in place)
  float* redp    = (float*)(Sc + 64 * 608);  // 64 x 4 partials
  float* rowstat = redp + 64 * 4;            // 64 x {max, inv_sum}

  const int t = blockIdx.x, h = blockIdx.y, b = blockIdx.z;
  const int s0 = t * 64;
  const int tid = threadIdx.x;
  const int lane = tid & 31, wave = tid >> 5;
  const int l = lane & 15, hi = lane >> 4;

  const size_t bh_off = (size_t)b * Ssz * DIMsz + (size_t)h * DHsz;   // + s*DIM + d

  // Q tile via async copy (q already scaled by 1/sqrt(DH) in projection)
  for (int i = tid; i < 64 * 8; i += 256) {
    int r = i >> 3, c4 = i & 7;
    async_cp16(Qs + i * 8, q + bh_off + (size_t)(s0 + r) * DIMsz + c4 * 8);
  }
  // Kext panel via async copy, branchless (clamped j)
  for (int i = tid; i < 608 * 8; i += 256) {
    int slot = i >> 3, c4 = i & 7;
    int j = (slot < 16) ? slot : (s0 - 256 + (slot - 16));
    j = (j < 0) ? 0 : ((j >= Ssz) ? (Ssz - 1) : j);
    async_cp16(KV + i * 8, k + bh_off + (size_t)j * DIMsz + c4 * 8);
  }
  async_wait0();
  __syncthreads();

  // Scores: 4 M-tiles x 38 N-tiles, round-robin over 8 waves (19 tiles each)
  for (int tile = wave; tile < 4 * 38; tile += 8) {
    int mt = tile / 38, nt = tile % 38;
    v8f acc = zero8();
    for (int ks = 0; ks < 2; ++ks) {
      v16h a  = load_a_frag(Qs + mt * 16 * 64 + ks * 32, 64, l, hi);
      v16h bb = load_bt_frag(KV + nt * 16 * 64 + ks * 32, 64, l, hi);
      acc = wmma32(a, bb, acc);
    }
    // Key validity depends only on the panel column -> hoist out of the r loop.
    int pcol = nt * 16 + l;
    int p = pcol - 16;
    int j = s0 - 256 + p;
    int jc = (j < 0) ? 0 : ((j >= Ssz) ? (Ssz - 1) : j);
    bool isSel = (pcol < 16);
    bool keyok = isSel ||
                 ((pcol < 592) && (j >= 0) && (j < Ssz) && (am[b * Ssz + jc] == 0));
    const _Float16 NEGH = (_Float16)(-__builtin_inff());
    for (int r = 0; r < 8; ++r) {
      int x = mt * 16 + hi * 8 + r;
      int m = p - x;                       // band offset, must be in [0, 512]
      bool valid = isSel || (keyok && (m >= 0) && (m <= 512));
      Sc[x * 608 + pcol] = valid ? (_Float16)acc[r] : NEGH;
    }
  }
  __syncthreads();

  // Softmax over 608 cols per row: 4 threads/row, each owns 152 contiguous cols
  {
    int row = tid >> 2, part = tid & 3;
    int4* rowp = (int4*)(Sc + row * 608 + part * 152);   // 19 x int4 (8 halfs each)
    float lmax = -3.0e38f;
    for (int c = 0; c < 19; ++c) {
      HI4 u; u.i4 = rowp[c];
      for (int e = 0; e < 8; ++e) lmax = fmaxf(lmax, (float)u.h[e]);
    }
    redp[row * 4 + part] = lmax;
    __syncthreads();
    if (part == 0)
      rowstat[row * 2] = fmaxf(fmaxf(redp[row * 4], redp[row * 4 + 1]),
                               fmaxf(redp[row * 4 + 2], redp[row * 4 + 3]));
    __syncthreads();
    float mx = rowstat[row * 2];
    float ls = 0.f;
    for (int c = 0; c < 19; ++c) {
      HI4 u; u.i4 = rowp[c];
      for (int e = 0; e < 8; ++e) {
        float ev = __expf((float)u.h[e] - mx);
        u.h[e] = (_Float16)ev;
        ls += ev;
      }
      rowp[c] = u.i4;
    }
    redp[row * 4 + part] = ls;
    __syncthreads();
    if (part == 0)
      rowstat[row * 2 + 1] =
          1.0f / (redp[row * 4] + redp[row * 4 + 1] + redp[row * 4 + 2] + redp[row * 4 + 3]);
    __syncthreads();
    float inv = rowstat[row * 2 + 1];
    for (int c = 0; c < 19; ++c) {
      HI4 u; u.i4 = rowp[c];
      for (int e = 0; e < 8; ++e) u.h[e] = (_Float16)((float)u.h[e] * inv);
      rowp[c] = u.i4;
    }
  }
  __syncthreads();

  // Reload KV as V-transposed: VT[d][slot], stride 608 (vector global read, scatter store)
  for (int i = tid; i < 608 * 8; i += 256) {
    int slot = i >> 3, dg = i & 7;
    int j = (slot < 16) ? slot : (s0 - 256 + (slot - 16));
    j = (j < 0) ? 0 : ((j >= Ssz) ? (Ssz - 1) : j);
    HI4 u; u.i4 = *(const int4*)(v + bh_off + (size_t)j * DIMsz + dg * 8);
    for (int e = 0; e < 8; ++e) KV[(dg * 8 + e) * 608 + slot] = u.h[e];
  }
  __syncthreads();

  // Output: O(64x64) = P(64x608) @ V(608x64); 16 tiles, 2 per wave, 19 K-steps
  for (int tt = 0; tt < 2; ++tt) {
    int tile = wave * 2 + tt;
    int mt = tile >> 2, nt = tile & 3;
    v8f acc = zero8();
    for (int ks = 0; ks < 19; ++ks) {
      v16h a  = load_a_frag(Sc + mt * 16 * 608 + ks * 32, 608, l, hi);
      v16h bb = load_bt_frag(KV + nt * 16 * 608 + ks * 32, 608, l, hi);
      acc = wmma32(a, bb, acc);
    }
    int d = nt * 16 + l;
    float* op = out + bh_off + (size_t)(s0 + mt * 16 + hi * 8) * DIMsz + d;
    for (int r = 0; r < 8; ++r)
      op[(size_t)r * DIMsz] = acc[r];
  }
}

// ---------------------------------------------------------------------------
// Global-token attention: out[:, :G] = softmax(qg.kg^T / sqrt(DH)) @ vg
// One block per (b, h); qg already carries the 1/sqrt(DH) scale.
// ---------------------------------------------------------------------------
__global__ __launch_bounds__(256) void global_attn(
    const _Float16* __restrict__ qg, const _Float16* __restrict__ kg,
    const _Float16* __restrict__ vg, float* __restrict__ out)
{
  extern __shared__ char smem[];
  _Float16* Qg  = (_Float16*)smem;         // 16 x 64
  _Float16* Sg  = Qg + 16 * 64;            // 16 x 2048 scores -> probs
  _Float16* KVc = Sg + 16 * 2048;          // K chunk 128x64 / VT chunk 64x128
  float* redp    = (float*)(KVc + 128 * 64);  // 16 x 16
  float* rowstat = redp + 16 * 16;            // 16 x {max, inv_sum}

  const int h = blockIdx.x, b = blockIdx.y;
  const int tid = threadIdx.x;
  const int lane = tid & 31, wave = tid >> 5;
  const int l = lane & 15, hi = lane >> 4;

  const size_t bh_off = (size_t)b * Ssz * DIMsz + (size_t)h * DHsz;
  const size_t qg_off = (size_t)b * Gsz * DIMsz + (size_t)h * DHsz;

  for (int i = tid; i < 16 * 8; i += 256) {
    int g = i >> 3, c4 = i & 7;
    async_cp16(Qg + i * 8, qg + qg_off + (size_t)g * DIMsz + c4 * 8);
  }
  async_wait0();
  __syncthreads();

  // Score phase: 16 chunks of 128 keys; 8 N-tiles -> one per wave
  for (int ch = 0; ch < 16; ++ch) {
    for (int i = tid; i < 128 * 8; i += 256) {
      int slot = i >> 3, c4 = i & 7;
      async_cp16(KVc + i * 8, kg + bh_off + (size_t)(ch * 128 + slot) * DIMsz + c4 * 8);
    }
    async_wait0();
    __syncthreads();
    {
      int nt = wave;
      v8f acc = zero8();
      for (int ks = 0; ks < 2; ++ks) {
        v16h a  = load_a_frag(Qg + ks * 32, 64, l, hi);
        v16h bb = load_bt_frag(KVc + nt * 16 * 64 + ks * 32, 64, l, hi);
        acc = wmma32(a, bb, acc);
      }
      int col = ch * 128 + nt * 16 + l;
      for (int r = 0; r < 8; ++r) {
        int g = hi * 8 + r;
        Sg[g * 2048 + col] = (_Float16)acc[r];
      }
    }
    __syncthreads();
  }

  // Softmax over 2048 cols per row: 16 threads/row, each owns 128 contiguous cols
  {
    int row = tid >> 4, part = tid & 15;
    int4* rowp = (int4*)(Sg + row * 2048 + part * 128);   // 16 x int4
    float lmax = -3.0e38f;
    for (int c = 0; c < 16; ++c) {
      HI4 u; u.i4 = rowp[c];
      for (int e = 0; e < 8; ++e) lmax = fmaxf(lmax, (float)u.h[e]);
    }
    redp[row * 16 + part] = lmax;
    __syncthreads();
    if (part == 0) {
      float mx = redp[row * 16];
      for (int u = 1; u < 16; ++u) mx = fmaxf(mx, redp[row * 16 + u]);
      rowstat[row * 2] = mx;
    }
    __syncthreads();
    float mx = rowstat[row * 2];
    float ls = 0.f;
    for (int c = 0; c < 16; ++c) {
      HI4 u; u.i4 = rowp[c];
      for (int e = 0; e < 8; ++e) {
        float ev = __expf((float)u.h[e] - mx);
        u.h[e] = (_Float16)ev;
        ls += ev;
      }
      rowp[c] = u.i4;
    }
    redp[row * 16 + part] = ls;
    __syncthreads();
    if (part == 0) {
      float s = 0.f;
      for (int u = 0; u < 16; ++u) s += redp[row * 16 + u];
      rowstat[row * 2 + 1] = 1.0f / s;
    }
    __syncthreads();
    float inv = rowstat[row * 2 + 1];
    for (int c = 0; c < 16; ++c) {
      HI4 u; u.i4 = rowp[c];
      for (int e = 0; e < 8; ++e) u.h[e] = (_Float16)((float)u.h[e] * inv);
      rowp[c] = u.i4;
    }
  }
  __syncthreads();

  // Output: O(16x64) = P(16x2048) @ Vg(2048x64), accumulated across chunks
  v8f acc = zero8();
  for (int ch = 0; ch < 16; ++ch) {
    for (int i = tid; i < 128 * 8; i += 256) {
      int slot = i >> 3, dg = i & 7;
      HI4 u; u.i4 = *(const int4*)(vg + bh_off + (size_t)(ch * 128 + slot) * DIMsz + dg * 8);
      for (int e = 0; e < 8; ++e) KVc[(dg * 8 + e) * 128 + slot] = u.h[e];
    }
    __syncthreads();
    if (wave < 4) {
      int nt = wave;
      for (int ks = 0; ks < 4; ++ks) {
        v16h a  = load_a_frag(Sg + ch * 128 + ks * 32, 2048, l, hi);
        v16h bb = load_bt_frag(KVc + nt * 16 * 128 + ks * 32, 128, l, hi);
        acc = wmma32(a, bb, acc);
      }
    }
    __syncthreads();
  }
  if (wave < 4) {
    int d = wave * 16 + l;
    float* op = out + bh_off + (size_t)(hi * 8) * DIMsz + d;
    for (int r = 0; r < 8; ++r)
      op[(size_t)r * DIMsz] = acc[r];
  }
}

__global__ void gather_hg(const float* __restrict__ hidden, float* __restrict__ hg) {
  int i = blockIdx.x * 256 + threadIdx.x;
  if (i < Bsz * Gsz * DIMsz) {
    int d = i % DIMsz;
    int row = i / DIMsz;          // b*G + g
    int b = row / Gsz, g = row % Gsz;
    hg[i] = hidden[((size_t)b * Ssz + g) * DIMsz + d];
  }
}

extern "C" void kernel_launch(void* const* d_in, const int* in_sizes, int n_in,
                              void* d_out, int out_size, void* d_ws, size_t ws_size,
                              hipStream_t stream) {
  (void)in_sizes; (void)n_in; (void)out_size; (void)ws_size;
  const float* hidden = (const float*)d_in[0];
  const int*   am     = (const int*)d_in[1];
  const float* Wq  = (const float*)d_in[2];
  const float* bq  = (const float*)d_in[3];
  const float* Wk  = (const float*)d_in[4];
  const float* bk  = (const float*)d_in[5];
  const float* Wv  = (const float*)d_in[6];
  const float* bv  = (const float*)d_in[7];
  const float* Wqg = (const float*)d_in[8];
  const float* bqg = (const float*)d_in[9];
  const float* Wkg = (const float*)d_in[10];
  const float* bkg = (const float*)d_in[11];
  const float* Wvg = (const float*)d_in[12];
  const float* bvg = (const float*)d_in[13];
  float* out = (float*)d_out;

  char* ws = (char*)d_ws;
  const size_t NTOK = (size_t)Bsz * Ssz;
  const size_t MAT  = NTOK * DIMsz;
  _Float16* q_h  = (_Float16*)ws; ws += MAT * 2;
  _Float16* k_h  = (_Float16*)ws; ws += MAT * 2;
  _Float16* v_h  = (_Float16*)ws; ws += MAT * 2;
  _Float16* kg_h = (_Float16*)ws; ws += MAT * 2;
  _Float16* vg_h = (_Float16*)ws; ws += MAT * 2;
  _Float16* qg_h = (_Float16*)ws; ws += (size_t)Bsz * Gsz * DIMsz * 2;
  float*    hgbuf = (float*)ws;   ws += (size_t)Bsz * Gsz * DIMsz * 4;

  gather_hg<<<(Bsz * Gsz * DIMsz + 255) / 256, 256, 0, stream>>>(hidden, hgbuf);

  const float qscale = 1.0f / sqrtf((float)DHsz);
  dim3 gg(DIMsz / 64, (unsigned)((NTOK + 127) / 128));
  gemm_bias_f16<<<gg, 256, 0, stream>>>(hidden, Wq,  bq,  q_h,  (int)NTOK, qscale);
  gemm_bias_f16<<<gg, 256, 0, stream>>>(hidden, Wk,  bk,  k_h,  (int)NTOK, 1.0f);
  gemm_bias_f16<<<gg, 256, 0, stream>>>(hidden, Wv,  bv,  v_h,  (int)NTOK, 1.0f);
  gemm_bias_f16<<<gg, 256, 0, stream>>>(hidden, Wkg, bkg, kg_h, (int)NTOK, 1.0f);
  gemm_bias_f16<<<gg, 256, 0, stream>>>(hidden, Wvg, bvg, vg_h, (int)NTOK, 1.0f);
  dim3 gq(DIMsz / 64, 1);
  gemm_bias_f16<<<gq, 256, 0, stream>>>(hgbuf, Wqg, bqg, qg_h, Bsz * Gsz, qscale);

  size_t band_smem = (size_t)(64 * 64 + 608 * 64 + 64 * 608) * 2 + (64 * 4 + 64 * 2) * 4;
  band_attn<<<dim3(Ssz / 64, Hsz, Bsz), 256, band_smem, stream>>>(q_h, k_h, v_h, am, out);

  size_t glob_smem = (size_t)(16 * 64 + 16 * 2048 + 128 * 64) * 2 + (16 * 16 + 16 * 2) * 4;
  global_attn<<<dim3(Hsz, Bsz), 256, glob_smem, stream>>>(qg_h, kg_h, vg_h, out);
}
